// HATBlock_19189913879197
// MI455X (gfx1250) — matrix operand order repaired and checked by reading, not compiled, not run on hardware
//
#include <hip/hip_runtime.h>
#include <hip/hip_bf16.h>

typedef __attribute__((ext_vector_type(16))) _Float16 v16h;
typedef __attribute__((ext_vector_type(8)))  _Float16 v8h;
typedef __attribute__((ext_vector_type(8)))  float    v8f;
typedef unsigned int v4u __attribute__((ext_vector_type(4)));
typedef int          v8i __attribute__((ext_vector_type(8)));
typedef int          v4i __attribute__((ext_vector_type(4)));

#define NTOK  36864     // 192*192
#define IMG   192
#define CDIM  180
#define NHEAD 6
#define HDIM  30
#define NWIN  144       // 12x12 windows of 16x16
#define HIDD  360
#define CCD   60
#define CSED  6
#define QKSCALE 0.18257418583505536f   // 30^-0.5

// Tensor Data Mover availability (5-arg on ROCm7.2/clang-22, 6-arg on clang-23)
#if defined(__has_builtin)
#  if __has_builtin(__builtin_amdgcn_tensor_load_to_lds) && \
      __has_builtin(__builtin_amdgcn_s_wait_tensorcnt)
#    define USE_TDM 1
#  endif
#endif
#ifndef USE_TDM
#  define USE_TDM 0
#endif

__device__ __forceinline__ v16h make_v16(v8h a, v8h b) {
  return __builtin_shufflevector(a, b, 0,1,2,3,4,5,6,7,8,9,10,11,12,13,14,15);
}

#if USE_TDM
// Issue a 2D TDM load: tile (rows x cols) of f32 from row-major tensor into LDS
// (packed rows, cols*4 bytes per row). OOB rows/cols return zero per ISA 8.2.
__device__ __forceinline__ void tdm_load_2d(const void* gptr, unsigned lds_off,
                                            unsigned rows, unsigned cols,
                                            unsigned tens_d0, unsigned tens_d1,
                                            unsigned stride_elems)
{
  unsigned long long ga = (unsigned long long)(uintptr_t)gptr;
  v4u g0;
  g0[0] = 1u;                                        // count=1, user D#
  g0[1] = lds_off;                                   // lds_addr (bytes)
  g0[2] = (unsigned)(ga & 0xffffffffu);              // global_addr[31:0]
  g0[3] = (unsigned)((ga >> 32) & 0x01ffffffu) | (2u << 30);  // addr[56:32]|type=2
  v8i g1;
  g1[0] = (int)(2u << 16);                           // mask=0, data_size=4B
  g1[1] = (int)(tens_d0 << 16);                      // tensor_dim0[15:0]
  g1[2] = (int)((tens_d0 >> 16) | (tens_d1 << 16));  // dim0[31:16] | dim1[15:0]
  g1[3] = (int)((tens_d1 >> 16) | (cols << 16));     // dim1[31:16] | tile_dim0
  g1[4] = (int)(rows & 0xffffu);                     // tile_dim1, tile_dim2=0
  g1[5] = (int)stride_elems;                         // tensor_dim0_stride[31:0]
  g1[6] = 0;
  g1[7] = 0;
  v4i gz = {0, 0, 0, 0};
#if __clang_major__ >= 23
  v8i gz8 = {0, 0, 0, 0, 0, 0, 0, 0};
  __builtin_amdgcn_tensor_load_to_lds(g0, g1, gz, gz, gz8, 0);
#else
  __builtin_amdgcn_tensor_load_to_lds(g0, g1, gz, gz, 0);
#endif
}
#endif

// ---------------------------------------------------------------------------
// LayerNorm over C=180: one wave per row, 8 rows per 256-thread block.
// ---------------------------------------------------------------------------
__global__ __launch_bounds__(256)
void ln_kernel(const float* __restrict__ in, const float* __restrict__ w,
               const float* __restrict__ b, float* __restrict__ out)
{
  const int lane = threadIdx.x & 31, wv = threadIdx.x >> 5;
  const int row = blockIdx.x * 8 + wv;
  const float* ip = in + (size_t)row * CDIM;
  float v[6]; float s = 0.f;
  #pragma unroll
  for (int i = 0; i < 6; ++i) {
    int d = lane + i * 32;
    v[i] = (d < CDIM) ? ip[d] : 0.f;
    s += v[i];
  }
  #pragma unroll
  for (int off = 16; off; off >>= 1) s += __shfl_xor(s, off, 32);
  float mu = s * (1.f / CDIM);
  float q = 0.f;
  #pragma unroll
  for (int i = 0; i < 6; ++i) {
    int d = lane + i * 32;
    float c = (d < CDIM) ? (v[i] - mu) : 0.f;
    q += c * c;
  }
  #pragma unroll
  for (int off = 16; off; off >>= 1) q += __shfl_xor(q, off, 32);
  float rstd = rsqrtf(q * (1.f / CDIM) + 1e-5f);
  float* op = out + (size_t)row * CDIM;
  #pragma unroll
  for (int i = 0; i < 6; ++i) {
    int d = lane + i * 32;
    if (d < CDIM) op[d] = (v[i] - mu) * rstd * w[d] + b[d];
  }
}

// ---------------------------------------------------------------------------
// Generic GEMM: out[M,N] = act(A[M,K] @ W[K,N] + bias) (+ residual)
// TDM (tensor_load_to_lds) stages f32 tiles into LDS; workgroup converts to
// f16 WMMA tiles (B transposed to [n][k]); v_wmma_f32_16x16x32_f16 compute.
// Block tile 64x64, 8 waves, wave tile 16x32 (A fragment reused).
// ACT: 0 = none, 1 = exact GELU.
// ---------------------------------------------------------------------------
template<int ACT>
__global__ __launch_bounds__(256)
void gemm_f16_wmma(const float* __restrict__ A, const float* __restrict__ W,
                   const float* __restrict__ bias, const float* __restrict__ residual,
                   float* __restrict__ out, int M, int K, int N)
{
  __shared__ _Float16 As[64][40];   // [m][k] f16, row pitch 80B (16B aligned)
  __shared__ _Float16 Bt[64][40];   // [n][k] f16 transposed
#if USE_TDM
  __shared__ float Ast[64 * 32];    // TDM staging, packed rows
  __shared__ float Bst[32 * 64];
#endif

  const int tid  = threadIdx.x;
  const int lane = tid & 31, wv = tid >> 5;
  const int nl = lane & 15, kh = lane >> 4;
  const int m0 = blockIdx.y * 64, n0 = blockIdx.x * 64;
  const int mt = wv & 3, nt = wv >> 2;

  v8f c0 = {}; v8f c1 = {};

  for (int k0 = 0; k0 < K; k0 += 32) {
#if USE_TDM
    if (wv == 0) {                      // one wave issues both DMA descriptors
      tdm_load_2d(A + (size_t)m0 * K + k0,
                  (unsigned)(uintptr_t)&Ast[0],
                  64u, 32u, (unsigned)(K - k0), (unsigned)(M - m0), (unsigned)K);
      tdm_load_2d(W + (size_t)k0 * N + n0,
                  (unsigned)(uintptr_t)&Bst[0],
                  32u, 64u, (unsigned)(N - n0), (unsigned)(K - k0), (unsigned)N);
      __builtin_amdgcn_s_wait_tensorcnt(0);
    }
    __syncthreads();                    // staged f32 tiles visible to all waves
    #pragma unroll
    for (int i = 0; i < 8; ++i) {       // convert A -> f16 tile
      int e = tid + i * 256;
      As[e >> 5][e & 31] = (_Float16)Ast[e];
    }
    #pragma unroll
    for (int i = 0; i < 8; ++i) {       // convert + transpose B -> f16 tile
      int e = tid + i * 256;
      Bt[e & 63][e >> 6] = (_Float16)Bst[e];
    }
#else
    if (k0 + 32 < K) {                  // gfx1250 prefetch of next K slice
      int pm = m0 + (tid >> 2);
      if (pm < M) __builtin_prefetch(&A[(size_t)pm * K + k0 + 32], 0, 1);
    }
    #pragma unroll
    for (int i = 0; i < 8; ++i) {       // A: 64x32 elems, coalesced
      int e = tid + i * 256;
      int r = e >> 5, kk = e & 31;
      int gm = m0 + r, gk = k0 + kk;
      float v = (gm < M && gk < K) ? A[(size_t)gm * K + gk] : 0.f;
      As[r][kk] = (_Float16)v;
    }
    #pragma unroll
    for (int i = 0; i < 8; ++i) {       // W: 32x64, coalesced over n, transposed
      int e = tid + i * 256;
      int kk = e >> 6, nn = e & 63;
      int gk = k0 + kk, gn = n0 + nn;
      float v = (gk < K && gn < N) ? W[(size_t)gk * N + gn] : 0.f;
      Bt[nn][kk] = (_Float16)v;
    }
#endif
    __syncthreads();

    // A fragment: m = lane&15; K(e) = 16*kh + e
    const _Float16* ar = &As[mt * 16 + nl][0];
    v16h af = make_v16(*(const v8h*)(ar + kh * 16), *(const v8h*)(ar + kh * 16 + 8));
    // B fragments: n = lane&15 (+16); K(e) = 16*kh + e
    const _Float16* b0r = &Bt[nt * 32 + nl][0];
    v16h bf0 = make_v16(*(const v8h*)(b0r + kh * 16), *(const v8h*)(b0r + kh * 16 + 8));
    const _Float16* b1r = &Bt[nt * 32 + 16 + nl][0];
    v16h bf1 = make_v16(*(const v8h*)(b1r + kh * 16), *(const v8h*)(b1r + kh * 16 + 8));

    c0 = __builtin_amdgcn_wmma_f32_16x16x32_f16(false, af, false, bf0, (short)0, c0, false, false);
    c1 = __builtin_amdgcn_wmma_f32_16x16x32_f16(false, af, false, bf1, (short)0, c1, false, false);
    __syncthreads();
  }

  // Epilogue (C layout: VGPR r -> m = r + 8*(lane>=16), n = lane&15)
  #pragma unroll
  for (int r = 0; r < 8; ++r) {
    int gm = m0 + mt * 16 + kh * 8 + r;
    if (gm >= M) continue;
    size_t base = (size_t)gm * N;
    int gc0 = n0 + nt * 32 + nl;
    int gc1 = gc0 + 16;
    if (gc0 < N) {
      float v = c0[r] + bias[gc0];
      if (ACT == 1) v = 0.5f * v * (1.f + erff(v * 0.70710678118654752f));
      if (residual) v += residual[base + gc0];
      out[base + gc0] = v;
    }
    if (gc1 < N) {
      float v = c1[r] + bias[gc1];
      if (ACT == 1) v = 0.5f * v * (1.f + erff(v * 0.70710678118654752f));
      if (residual) v += residual[base + gc1];
      out[base + gc1] = v;
    }
  }
}

// ---------------------------------------------------------------------------
// Fused window MSA (flash style). One block per (window, head).
// qkv: [NTOK][540] token-major (q|k|v, each [head][30]).
// Handles cyclic shift via gather/scatter indices; rel bias + shift mask.
// ---------------------------------------------------------------------------
__global__ __launch_bounds__(256)
void win_msa_wmma(const float* __restrict__ qkv, const float* __restrict__ relb,
                  float* __restrict__ outp, int shift)
{
  __shared__ _Float16 Qs[256][40];      // q, scaled, f16
  __shared__ _Float16 Kc[32][40];       // key chunk [key][dim]
  __shared__ _Float16 Vc[32][40];       // value chunk transposed [dim][key]
  __shared__ _Float16 Pb[8][16][40];    // per-wave P scratch
  __shared__ float    Rb[961];          // rel-bias for this head
  __shared__ int      rowIdx[256];
  __shared__ unsigned char lab[256];

  const int tid = threadIdx.x, lane = tid & 31, wv = tid >> 5;
  const int nl = lane & 15, kh = lane >> 4;
  const int win = blockIdx.x, head = blockIdx.y;
  const int wy = win / 12, wx = win % 12;

  for (int i = tid; i < 961; i += 256) Rb[i] = relb[i * NHEAD + head];
  {
    int t = tid;                         // 256 tokens in window
    int ty = t >> 4, tx = t & 15;
    int hh = wy * 16 + ty, ww = wx * 16 + tx;   // rolled-frame coords
    int oh = hh + shift; if (oh >= IMG) oh -= IMG;
    int ow = ww + shift; if (ow >= IMG) ow -= IMG;
    int row = oh * IMG + ow;
    rowIdx[t] = row;
    int lh = hh < 176 ? 0 : (hh < 184 ? 1 : 2);
    int lw = ww < 176 ? 0 : (ww < 184 ? 1 : 2);
    lab[t] = (unsigned char)(lh * 3 + lw);
    const float* qp = qkv + (size_t)row * 540 + head * HDIM;
    #pragma unroll
    for (int d = 0; d < 32; ++d)
      Qs[t][d] = (d < HDIM) ? (_Float16)(qp[d] * QKSCALE) : (_Float16)0.f;
  }
  __syncthreads();

  v16h qf[2];
  #pragma unroll
  for (int qi = 0; qi < 2; ++qi) {
    const _Float16* ar = &Qs[(wv + qi * 8) * 16 + nl][0];
    qf[qi] = make_v16(*(const v8h*)(ar + kh * 16), *(const v8h*)(ar + kh * 16 + 8));
  }
  float mrun[2][8], lrun[2][8];
  v8f o0[2], o1[2];
  #pragma unroll
  for (int qi = 0; qi < 2; ++qi) {
    o0[qi] = (v8f){}; o1[qi] = (v8f){};
    #pragma unroll
    for (int r = 0; r < 8; ++r) { mrun[qi][r] = -1e30f; lrun[qi][r] = 0.f; }
  }

  for (int ch = 0; ch < 8; ++ch) {
    #pragma unroll
    for (int i = 0; i < 4; ++i) {        // load 32-key K/V chunk
      int e = tid + i * 256;
      int key = e >> 5, d = e & 31;
      int row = rowIdx[ch * 32 + key];
      float kk = 0.f, vvv = 0.f;
      if (d < HDIM) {
        const float* base = qkv + (size_t)row * 540;
        kk  = base[180 + head * HDIM + d];
        vvv = base[360 + head * HDIM + d];
      }
      Kc[key][d] = (_Float16)kk;
      Vc[d][key] = (_Float16)vvv;
    }
    __syncthreads();

    #pragma unroll
    for (int qi = 0; qi < 2; ++qi) {
      const int qt = wv + qi * 8;
      v8f s0 = {}; v8f s1 = {};
      const _Float16* b0 = &Kc[nl][0];
      v16h kf0 = make_v16(*(const v8h*)(b0 + kh * 16), *(const v8h*)(b0 + kh * 16 + 8));
      const _Float16* b1 = &Kc[16 + nl][0];
      v16h kf1 = make_v16(*(const v8h*)(b1 + kh * 16), *(const v8h*)(b1 + kh * 16 + 8));
      s0 = __builtin_amdgcn_wmma_f32_16x16x32_f16(false, qf[qi], false, kf0, (short)0, s0, false, false);
      s1 = __builtin_amdgcn_wmma_f32_16x16x32_f16(false, qf[qi], false, kf1, (short)0, s1, false, false);

      float sv0[8], sv1[8];
      #pragma unroll
      for (int r = 0; r < 8; ++r) {
        int qrow = qt * 16 + kh * 8 + r;
        int kc0 = ch * 32 + nl, kc1 = kc0 + 16;
        int qy = qrow >> 4, qx = qrow & 15;
        float bb0 = Rb[(qy - (kc0 >> 4) + 15) * 31 + (qx - (kc0 & 15) + 15)];
        float bb1 = Rb[(qy - (kc1 >> 4) + 15) * 31 + (qx - (kc1 & 15) + 15)];
        float mk0 = 0.f, mk1 = 0.f;
        if (shift > 0) {
          if (lab[qrow] != lab[kc0]) mk0 = -100.f;
          if (lab[qrow] != lab[kc1]) mk1 = -100.f;
        }
        sv0[r] = s0[r] + bb0 + mk0;
        sv1[r] = s1[r] + bb1 + mk1;
      }
      #pragma unroll
      for (int r = 0; r < 8; ++r) {       // online softmax per q row
        float mx = fmaxf(sv0[r], sv1[r]);
        #pragma unroll
        for (int off = 8; off; off >>= 1) mx = fmaxf(mx, __shfl_xor(mx, off, 16));
        float mnew  = fmaxf(mrun[qi][r], mx);
        float alpha = __expf(mrun[qi][r] - mnew);
        float p0 = __expf(sv0[r] - mnew);
        float p1 = __expf(sv1[r] - mnew);
        float ps = p0 + p1;
        #pragma unroll
        for (int off = 8; off; off >>= 1) ps += __shfl_xor(ps, off, 16);
        lrun[qi][r] = lrun[qi][r] * alpha + ps;
        mrun[qi][r] = mnew;
        o0[qi][r] *= alpha; o1[qi][r] *= alpha;
        Pb[wv][kh * 8 + r][nl]      = (_Float16)p0;
        Pb[wv][kh * 8 + r][16 + nl] = (_Float16)p1;
      }
      const _Float16* pr = &Pb[wv][nl][0];  // wave-local, DS in-order
      v16h pf = make_v16(*(const v8h*)(pr + kh * 16), *(const v8h*)(pr + kh * 16 + 8));
      const _Float16* v0r = &Vc[nl][0];
      v16h vf0 = make_v16(*(const v8h*)(v0r + kh * 16), *(const v8h*)(v0r + kh * 16 + 8));
      const _Float16* v1r = &Vc[16 + nl][0];
      v16h vf1 = make_v16(*(const v8h*)(v1r + kh * 16), *(const v8h*)(v1r + kh * 16 + 8));
      o0[qi] = __builtin_amdgcn_wmma_f32_16x16x32_f16(false, pf, false, vf0, (short)0, o0[qi], false, false);
      o1[qi] = __builtin_amdgcn_wmma_f32_16x16x32_f16(false, pf, false, vf1, (short)0, o1[qi], false, false);
    }
    __syncthreads();
  }

  #pragma unroll
  for (int qi = 0; qi < 2; ++qi) {
    int qt = wv + qi * 8;
    #pragma unroll
    for (int r = 0; r < 8; ++r) {
      int qrow = qt * 16 + kh * 8 + r;
      int row = rowIdx[qrow];
      float inv = 1.f / lrun[qi][r];
      float* op = outp + (size_t)row * CDIM + head * HDIM;
      op[nl] = o0[qi][r] * inv;
      if (nl + 16 < HDIM) op[nl + 16] = o1[qi][r] * inv;
    }
  }
}

// ---------------------------------------------------------------------------
// OCAB attention: q over 16x16 windows, k/v over overlapping 24x24 patches
// (576 keys, 18 chunks). OOB patch rows = kv bias row (pad applied pre-proj).
// ---------------------------------------------------------------------------
__global__ __launch_bounds__(256)
void ocab_attn_wmma(const float* __restrict__ qb, const float* __restrict__ kv,
                    const float* __restrict__ kvbias, float* __restrict__ outp)
{
  __shared__ _Float16 Qs[256][40];
  __shared__ _Float16 Kc[32][40];
  __shared__ _Float16 Vc[32][40];
  __shared__ _Float16 Pb[8][16][40];
  __shared__ int qRow[256];
  __shared__ int kvRow[576];

  const int tid = threadIdx.x, lane = tid & 31, wv = tid >> 5;
  const int nl = lane & 15, kh = lane >> 4;
  const int win = blockIdx.x, head = blockIdx.y;
  const int wy = win / 12, wx = win % 12;

  {
    int t = tid;
    int row = (wy * 16 + (t >> 4)) * IMG + wx * 16 + (t & 15);
    qRow[t] = row;
    const float* qp = qb + (size_t)row * CDIM + head * HDIM;
    #pragma unroll
    for (int d = 0; d < 32; ++d)
      Qs[t][d] = (d < HDIM) ? (_Float16)(qp[d] * QKSCALE) : (_Float16)0.f;
  }
  for (int t = tid; t < 576; t += 256) {
    int hh = wy * 16 - 4 + t / 24, ww = wx * 16 - 4 + t % 24;
    kvRow[t] = (hh >= 0 && hh < IMG && ww >= 0 && ww < IMG) ? hh * IMG + ww : -1;
  }
  __syncthreads();

  v16h qf[2];
  #pragma unroll
  for (int qi = 0; qi < 2; ++qi) {
    const _Float16* ar = &Qs[(wv + qi * 8) * 16 + nl][0];
    qf[qi] = make_v16(*(const v8h*)(ar + kh * 16), *(const v8h*)(ar + kh * 16 + 8));
  }
  float mrun[2][8], lrun[2][8];
  v8f o0[2], o1[2];
  #pragma unroll
  for (int qi = 0; qi < 2; ++qi) {
    o0[qi] = (v8f){}; o1[qi] = (v8f){};
    #pragma unroll
    for (int r = 0; r < 8; ++r) { mrun[qi][r] = -1e30f; lrun[qi][r] = 0.f; }
  }

  for (int ch = 0; ch < 18; ++ch) {
    #pragma unroll
    for (int i = 0; i < 4; ++i) {
      int e = tid + i * 256;
      int key = e >> 5, d = e & 31;
      int row = kvRow[ch * 32 + key];
      float kk = 0.f, vvv = 0.f;
      if (d < HDIM) {
        if (row >= 0) {
          const float* base = kv + (size_t)row * 360;
          kk  = base[head * HDIM + d];
          vvv = base[180 + head * HDIM + d];
        } else {
          kk  = kvbias[head * HDIM + d];
          vvv = kvbias[180 + head * HDIM + d];
        }
      }
      Kc[key][d] = (_Float16)kk;
      Vc[d][key] = (_Float16)vvv;
    }
    __syncthreads();

    #pragma unroll
    for (int qi = 0; qi < 2; ++qi) {
      v8f s0 = {}; v8f s1 = {};
      const _Float16* b0 = &Kc[nl][0];
      v16h kf0 = make_v16(*(const v8h*)(b0 + kh * 16), *(const v8h*)(b0 + kh * 16 + 8));
      const _Float16* b1 = &Kc[16 + nl][0];
      v16h kf1 = make_v16(*(const v8h*)(b1 + kh * 16), *(const v8h*)(b1 + kh * 16 + 8));
      s0 = __builtin_amdgcn_wmma_f32_16x16x32_f16(false, qf[qi], false, kf0, (short)0, s0, false, false);
      s1 = __builtin_amdgcn_wmma_f32_16x16x32_f16(false, qf[qi], false, kf1, (short)0, s1, false, false);

      #pragma unroll
      for (int r = 0; r < 8; ++r) {
        float mx = fmaxf(s0[r], s1[r]);
        #pragma unroll
        for (int off = 8; off; off >>= 1) mx = fmaxf(mx, __shfl_xor(mx, off, 16));
        float mnew  = fmaxf(mrun[qi][r], mx);
        float alpha = __expf(mrun[qi][r] - mnew);
        float p0 = __expf(s0[r] - mnew);
        float p1 = __expf(s1[r] - mnew);
        float ps = p0 + p1;
        #pragma unroll
        for (int off = 8; off; off >>= 1) ps += __shfl_xor(ps, off, 16);
        lrun[qi][r] = lrun[qi][r] * alpha + ps;
        mrun[qi][r] = mnew;
        o0[qi][r] *= alpha; o1[qi][r] *= alpha;
        Pb[wv][kh * 8 + r][nl]      = (_Float16)p0;
        Pb[wv][kh * 8 + r][16 + nl] = (_Float16)p1;
      }
      const _Float16* pr = &Pb[wv][nl][0];
      v16h pf = make_v16(*(const v8h*)(pr + kh * 16), *(const v8h*)(pr + kh * 16 + 8));
      const _Float16* v0r = &Vc[nl][0];
      v16h vf0 = make_v16(*(const v8h*)(v0r + kh * 16), *(const v8h*)(v0r + kh * 16 + 8));
      const _Float16* v1r = &Vc[16 + nl][0];
      v16h vf1 = make_v16(*(const v8h*)(v1r + kh * 16), *(const v8h*)(v1r + kh * 16 + 8));
      o0[qi] = __builtin_amdgcn_wmma_f32_16x16x32_f16(false, pf, false, vf0, (short)0, o0[qi], false, false);
      o1[qi] = __builtin_amdgcn_wmma_f32_16x16x32_f16(false, pf, false, vf1, (short)0, o1[qi], false, false);
    }
    __syncthreads();
  }

  #pragma unroll
  for (int qi = 0; qi < 2; ++qi) {
    int qt = wv + qi * 8;
    #pragma unroll
    for (int r = 0; r < 8; ++r) {
      int row = qRow[qt * 16 + kh * 8 + r];
      float inv = 1.f / lrun[qi][r];
      float* op = outp + (size_t)row * CDIM + head * HDIM;
      op[nl] = o0[qi][r] * inv;
      if (nl + 16 < HDIM) op[nl + 16] = o1[qi][r] * inv;
    }
  }
}

// ---------------------------------------------------------------------------
// Direct 3x3 conv on token-major [HW][Cin] -> [HW][Cout], zero pad.
// Optional per-Cin gate folded into weights, optional residual.
// Grid: (HW/256, Cout); weights for cout staged in LDS.
// ---------------------------------------------------------------------------
__global__ __launch_bounds__(256)
void conv3x3_kernel(const float* __restrict__ in, const float* __restrict__ w,
                    const float* __restrict__ bias, const float* __restrict__ scale,
                    const float* __restrict__ residual, float* __restrict__ out,
                    int Cin, int Cout)
{
  __shared__ float Wsh[CDIM * 9];
  const int cout = blockIdx.y;
  for (int i = threadIdx.x; i < Cin * 9; i += 256) {
    float sc = scale ? scale[i / 9] : 1.f;
    Wsh[i] = w[(size_t)cout * Cin * 9 + i] * sc;
  }
  __syncthreads();
  const int pix = blockIdx.x * 256 + threadIdx.x;
  const int py = pix / IMG, px = pix % IMG;
  float acc = bias[cout];
  #pragma unroll
  for (int dy = -1; dy <= 1; ++dy) {
    int yy = py + dy; if (yy < 0 || yy >= IMG) continue;
    #pragma unroll
    for (int dx = -1; dx <= 1; ++dx) {
      int xx = px + dx; if (xx < 0 || xx >= IMG) continue;
      const float* ip = in + (size_t)(yy * IMG + xx) * Cin;
      const int tap = (dy + 1) * 3 + (dx + 1);
      float a = 0.f;
      for (int ci = 0; ci < Cin; ++ci) a += ip[ci] * Wsh[ci * 9 + tap];
      acc += a;
    }
  }
  float r = residual ? residual[(size_t)pix * Cout + cout] : 0.f;
  out[(size_t)pix * Cout + cout] = acc + r;
}

// Per-channel mean over all pixels (SE squeeze).
__global__ __launch_bounds__(256)
void colmean_kernel(const float* __restrict__ comp, float* __restrict__ s, int Ccc)
{
  const int c = blockIdx.x;
  float a = 0.f;
  for (int i = threadIdx.x; i < NTOK; i += 256) a += comp[(size_t)i * Ccc + c];
  __shared__ float red[256];
  red[threadIdx.x] = a; __syncthreads();
  for (int st = 128; st; st >>= 1) {
    if (threadIdx.x < st) red[threadIdx.x] += red[threadIdx.x + st];
    __syncthreads();
  }
  if (threadIdx.x == 0) s[c] = red[0] * (1.f / NTOK);
}

// SE gate: g = sigmoid(relu(s@w1+b1)@w2+b2). Single tiny block.
__global__ __launch_bounds__(64)
void se_gate_kernel(const float* __restrict__ s, const float* __restrict__ w1,
                    const float* __restrict__ b1, const float* __restrict__ w2,
                    const float* __restrict__ b2, float* __restrict__ g)
{
  __shared__ float t[CSED];
  const int tid = threadIdx.x;
  if (tid < CSED) {
    float a = b1[tid];
    for (int c = 0; c < CCD; ++c) a += s[c] * w1[c * CSED + tid];
    t[tid] = fmaxf(a, 0.f);
  }
  __syncthreads();
  if (tid < CCD) {
    float a = b2[tid];
    for (int j = 0; j < CSED; ++j) a += t[j] * w2[j * CCD + tid];
    g[tid] = 1.f / (1.f + __expf(-a));
  }
}

// ---------------------------------------------------------------------------
extern "C" void kernel_launch(void* const* d_in, const int* in_sizes, int n_in,
                              void* d_out, int out_size, void* d_ws, size_t ws_size,
                              hipStream_t stream)
{
  const float* x_in   = (const float*)d_in[0];
  const float* n1w    = (const float*)d_in[1];
  const float* n1b    = (const float*)d_in[2];
  const float* qkvw   = (const float*)d_in[3];
  const float* qkvb   = (const float*)d_in[4];
  const float* relb   = (const float*)d_in[5];
  const float* projw  = (const float*)d_in[6];
  const float* projb  = (const float*)d_in[7];
  const float* ccw    = (const float*)d_in[8];
  const float* ccb    = (const float*)d_in[9];
  const float* ca1w   = (const float*)d_in[10];
  const float* ca1b   = (const float*)d_in[11];
  const float* ca2w   = (const float*)d_in[12];
  const float* ca2b   = (const float*)d_in[13];
  const float* cew    = (const float*)d_in[14];
  const float* ceb    = (const float*)d_in[15];
  const float* n2w    = (const float*)d_in[16];
  const float* n2b    = (const float*)d_in[17];
  const float* m1w    = (const float*)d_in[18];
  const float* m1b    = (const float*)d_in[19];
  const float* m2w    = (const float*)d_in[20];
  const float* m2b    = (const float*)d_in[21];
  const float* onw    = (const float*)d_in[22];
  const float* onb    = (const float*)d_in[23];
  const float* oqw    = (const float*)d_in[24];
  const float* oqb    = (const float*)d_in[25];
  const float* okvw   = (const float*)d_in[26];
  const float* okvb   = (const float*)d_in[27];
  const float* opw    = (const float*)d_in[28];
  const float* opb    = (const float*)d_in[29];
  const float* convw  = (const float*)d_in[30];
  const float* convb  = (const float*)d_in[31];

  float* ws = (float*)d_ws;
  size_t off = 0;
  float* xA   = ws + off; off += (size_t)NTOK * CDIM;
  float* xB   = ws + off; off += (size_t)NTOK * CDIM;
  float* xn   = ws + off; off += (size_t)NTOK * CDIM;
  float* big  = ws + off; off += (size_t)NTOK * 540;   // qkv / (q|kv) / mlp hidden
  float* att  = ws + off; off += (size_t)NTOK * CDIM;
  float* comp = ws + off; off += (size_t)NTOK * CCD;
  float* svec = ws + off; off += 64;
  float* gvec = ws + off; off += 64;

  hipMemcpyAsync(xA, x_in, (size_t)NTOK * CDIM * sizeof(float),
                 hipMemcpyDeviceToDevice, stream);
  float* cur = xA; float* nxt = xB;
  const dim3 gLN(NTOK / 8), gAttn(NWIN, NHEAD);

  for (int i = 0; i < 6; ++i) {
    const int shift = (i & 1) ? 8 : 0;
    // LN1 + qkv
    ln_kernel<<<gLN, 256, 0, stream>>>(cur, n1w + i*CDIM, n1b + i*CDIM, xn);
    gemm_f16_wmma<0><<<dim3(9, NTOK/64), 256, 0, stream>>>(
        xn, qkvw + (size_t)i*CDIM*540, qkvb + i*540, nullptr, big, NTOK, CDIM, 540);
    // window MSA (fused flash)
    win_msa_wmma<<<gAttn, 256, 0, stream>>>(big, relb + (size_t)i*961*NHEAD, att, shift);
    // proj + residual
    gemm_f16_wmma<0><<<dim3(3, NTOK/64), 256, 0, stream>>>(
        att, projw + (size_t)i*CDIM*CDIM, projb + i*CDIM, cur, nxt, NTOK, CDIM, CDIM);
    { float* t = cur; cur = nxt; nxt = t; }
    // channel attention branch
    conv3x3_kernel<<<dim3(NTOK/256, CCD), 256, 0, stream>>>(
        cur, ccw + (size_t)i*CCD*CDIM*9, ccb + i*CCD, nullptr, nullptr, comp, CDIM, CCD);
    colmean_kernel<<<CCD, 256, 0, stream>>>(comp, svec, CCD);
    se_gate_kernel<<<1, 64, 0, stream>>>(svec, ca1w + i*CCD*CSED, ca1b + i*CSED,
                                         ca2w + i*CSED*CCD, ca2b + i*CCD, gvec);
    conv3x3_kernel<<<dim3(NTOK/256, CDIM), 256, 0, stream>>>(
        comp, cew + (size_t)i*CDIM*CCD*9, ceb + i*CDIM, gvec, cur, nxt, CCD, CDIM);
    { float* t = cur; cur = nxt; nxt = t; }
    // LN2 + MLP (GELU)
    ln_kernel<<<gLN, 256, 0, stream>>>(cur, n2w + i*CDIM, n2b + i*CDIM, xn);
    gemm_f16_wmma<1><<<dim3(6, NTOK/64), 256, 0, stream>>>(
        xn, m1w + (size_t)i*CDIM*HIDD, m1b + i*HIDD, nullptr, big, NTOK, CDIM, HIDD);
    gemm_f16_wmma<0><<<dim3(3, NTOK/64), 256, 0, stream>>>(
        big, m2w + (size_t)i*HIDD*CDIM, m2b + i*CDIM, cur, nxt, NTOK, HIDD, CDIM);
    { float* t = cur; cur = nxt; nxt = t; }
  }

  // OCAB
  float* qbuf  = big;
  float* kvbuf = big + (size_t)NTOK * CDIM;
  ln_kernel<<<gLN, 256, 0, stream>>>(cur, onw, onb, xn);
  gemm_f16_wmma<0><<<dim3(3, NTOK/64), 256, 0, stream>>>(
      xn, oqw, oqb, nullptr, qbuf, NTOK, CDIM, CDIM);
  gemm_f16_wmma<0><<<dim3(6, NTOK/64), 256, 0, stream>>>(
      xn, okvw, okvb, nullptr, kvbuf, NTOK, CDIM, 360);
  ocab_attn_wmma<<<gAttn, 256, 0, stream>>>(qbuf, kvbuf, okvb, att);
  gemm_f16_wmma<0><<<dim3(3, NTOK/64), 256, 0, stream>>>(
      att, opw, opb, cur, nxt, NTOK, CDIM, CDIM);
  { float* t = cur; cur = nxt; nxt = t; }

  // final conv + global shortcut -> d_out
  conv3x3_kernel<<<dim3(NTOK/256, CDIM), 256, 0, stream>>>(
      cur, convw, convb, nullptr, x_in, (float*)d_out, CDIM, CDIM);
}
